// SparseTransformerCrossBlock_30751965839835
// MI455X (gfx1250) — matrix operand order repaired and checked by reading, not compiled, not run on hardware
//
#include <hip/hip_runtime.h>
#include <hip/hip_bf16.h>
#include <math.h>

// ---------------------------------------------------------------------------
// Types for CDNA5 WMMA / TDM
// ---------------------------------------------------------------------------
typedef __attribute__((ext_vector_type(16))) __bf16 v16bf;
typedef __attribute__((ext_vector_type(8)))  float  v8f;
typedef __attribute__((ext_vector_type(4)))  unsigned int u32x4;
typedef __attribute__((ext_vector_type(8)))  int          i32x8;
typedef __attribute__((ext_vector_type(4)))  int          i32x4;

union AFrag {
    v16bf        v;
    unsigned int u[8];
};

static __device__ __forceinline__ v8f wmma_bf16(v16bf a, v16bf b, v8f c) {
    // D = A(16x32 bf16) * B(32x16 bf16) + C(16x16 f32)
    return __builtin_amdgcn_wmma_f32_16x16x32_bf16(
        /*neg_a=*/false, a, /*neg_b=*/false, b,
        /*c_mod=*/(short)0, c, /*reuse_a=*/false, /*reuse_b=*/false);
}

static __device__ __forceinline__ unsigned short f2bf(float f) {
    unsigned int x = __float_as_uint(f);
    unsigned int r = x + 0x7FFFu + ((x >> 16) & 1u);   // round-to-nearest-even
    return (unsigned short)(r >> 16);
}

static __device__ __forceinline__ float gelu_tanh(float x) {
    float x3 = x * x * x;
    return 0.5f * x * (1.0f + tanhf(0.7978845608028654f * (x + 0.044715f * x3)));
}

// ---------------------------------------------------------------------------
// Tensor Data Mover: async DMA of a 2D bf16 tile (tile_h rows x tile_w elems,
// row stride = row_stride elems) from global memory into LDS at lds_addr.
// D# descriptor per CDNA5 ISA 8.3/8.4 (groups 0/1; groups 2/3 zero => 2D).
// Issued once per wave; tracked by TENSORcnt (in-order within a wave).
// ---------------------------------------------------------------------------
static __device__ __forceinline__ void tdm_load_2d(
    unsigned lds_addr, const void* gptr,
    unsigned tile_w, unsigned tile_h,
    unsigned long long row_stride,
    unsigned tensor_w, unsigned tensor_h) {

    unsigned long long ga = (unsigned long long)gptr;

    u32x4 g0;
    g0[0] = 1u;                                   // count=1 (valid user D#)
    g0[1] = lds_addr;                             // LDS byte address
    g0[2] = (unsigned)ga;                         // global_addr[31:0]
    g0[3] = (unsigned)((ga >> 32) & 0x01FFFFFFu)  // global_addr[56:32]
            | 0x80000000u;                        // type=2 ("image")

    i32x8 g1;
    g1[0] = (int)(1u << 16);                      // data_size=1 -> 2 bytes
    g1[1] = (int)((tensor_w & 0xFFFFu) << 16);    // tensor_dim0[15:0]
    g1[2] = (int)((tensor_w >> 16) | ((tensor_h & 0xFFFFu) << 16)); // td0 hi | td1 lo
    g1[3] = (int)((tensor_h >> 16) | (tile_w << 16));               // td1 hi | tile_dim0
    g1[4] = (int)tile_h;                          // tile_dim1 (tile_dim2=0)
    g1[5] = (int)(unsigned)row_stride;            // tensor_dim0_stride[31:0]
    g1[6] = (int)(unsigned)(row_stride >> 32);    // tensor_dim0_stride[47:32]
    g1[7] = 0;
    i32x4 z4 = (i32x4){0, 0, 0, 0};
#if __clang_major__ >= 23
    i32x8 z8 = (i32x8){0, 0, 0, 0, 0, 0, 0, 0};
    __builtin_amdgcn_tensor_load_to_lds(g0, g1, z4, z4, z8, 0);
#else
    __builtin_amdgcn_tensor_load_to_lds(g0, g1, z4, z4, 0);
#endif
}

// ---------------------------------------------------------------------------
// f32 -> bf16 conversion (context / activations)
// ---------------------------------------------------------------------------
__global__ void cvt_f32_bf16_kernel(const float* __restrict__ in,
                                    unsigned short* __restrict__ out, size_t n) {
    size_t i = (size_t)blockIdx.x * blockDim.x + threadIdx.x;
    size_t stride = (size_t)gridDim.x * blockDim.x;
    for (; i < n; i += stride) out[i] = f2bf(in[i]);
}

// ---------------------------------------------------------------------------
// Transpose + convert: W (KxN f32, row-major) -> W^T (NxK bf16, row-major).
// Done once per weight per launch; makes the GEMM B tile a contiguous-stride
// 2D tile that the Tensor Data Mover can DMA directly in fragment layout.
// ---------------------------------------------------------------------------
__global__ __launch_bounds__(256) void cvtT_f32_bf16_kernel(
    const float* __restrict__ in, unsigned short* __restrict__ out, int K, int N) {
    __shared__ float t[32][33];
    int k0 = blockIdx.x * 32, n0 = blockIdx.y * 32;
    int tx = threadIdx.x & 31, ty = threadIdx.x >> 5;   // 32 x 8
#pragma unroll
    for (int i = 0; i < 32; i += 8)
        t[ty + i][tx] = in[(size_t)(k0 + ty + i) * N + n0 + tx];
    __syncthreads();
#pragma unroll
    for (int i = 0; i < 32; i += 8)
        out[(size_t)(n0 + ty + i) * K + k0 + tx] = f2bf(t[tx][ty + i]);
}

// ---------------------------------------------------------------------------
// LayerNorm over C=1024, bf16 output (feeds the next GEMM's A matrix)
// ---------------------------------------------------------------------------
__global__ __launch_bounds__(256) void ln_bf16_kernel(const float* __restrict__ x,
                                                      unsigned short* __restrict__ out) {
    const int C = 1024;
    size_t row = blockIdx.x;
    const float* xr = x + row * C;
    int t = threadIdx.x;

    float v0 = xr[t], v1 = xr[t + 256], v2 = xr[t + 512], v3 = xr[t + 768];
    float s  = v0 + v1 + v2 + v3;
    float s2 = v0 * v0 + v1 * v1 + v2 * v2 + v3 * v3;

    __shared__ float sh1[256], sh2[256];
    sh1[t] = s; sh2[t] = s2;
    __syncthreads();
    for (int off = 128; off; off >>= 1) {
        if (t < off) { sh1[t] += sh1[t + off]; sh2[t] += sh2[t + off]; }
        __syncthreads();
    }
    float mu   = sh1[0] * (1.0f / C);
    float var  = sh2[0] * (1.0f / C) - mu * mu;
    float rstd = rsqrtf(var + 1e-6f);

    unsigned short* orow = out + row * C;
    orow[t]       = f2bf((v0 - mu) * rstd);
    orow[t + 256] = f2bf((v1 - mu) * rstd);
    orow[t + 512] = f2bf((v2 - mu) * rstd);
    orow[t + 768] = f2bf((v3 - mu) * rstd);
}

// ---------------------------------------------------------------------------
// bf16 WMMA GEMM: C[M,N] = A[M,K] * W[K,N] + bias (+gelu) (+residual)
// W is passed pre-transposed (Wt: NxK row-major). Block: 256 thr = 8 waves.
// Block tile 128x128, BK=32. Wave grid 4(M) x 2(N); each wave 32x64 output.
// Both A and Wt tiles are DMA'd by the Tensor Data Mover (wave 0), double
// buffered: issue next tile pair, then s_wait_tensorcnt 2 for current pair.
// Main loop = TDM + ds_load_b128 fragments + 8x v_wmma. No global loads.
// ---------------------------------------------------------------------------
#define BM 128
#define BN 128
#define BK 32
#define TILE_BYTES (BM * BK * 2)   // 8192 (BM==BN)

__global__ __launch_bounds__(256) void gemm_bf16_kernel(
    const unsigned short* __restrict__ A, int lda,
    const unsigned short* __restrict__ Wt,          // N x K row-major
    const float* __restrict__ bias,
    const float* __restrict__ residual,
    float* __restrict__ outf,
    unsigned short* __restrict__ outb,
    int M, int N, int K, int ldc, int do_gelu) {

    __shared__ unsigned short As[2][BM * BK];   // [m][k] (TDM dest)
    __shared__ unsigned short Bs[2][BN * BK];   // [n][k] (TDM dest)

    const int tid  = threadIdx.x;
    const int wave = tid >> 5;
    const int lane = tid & 31;
    const int half = lane >> 4;
    const int ml   = lane & 15;

    const int wm0 = (wave & 3) * 32;   // wave row offset in block tile
    const int wn0 = (wave >> 2) * 64;  // wave col offset in block tile

    const int blockRow = blockIdx.x * BM;
    const int blockCol = blockIdx.y * BN;

    const unsigned lds_As = (unsigned)(unsigned long long)&As[0][0];
    const unsigned lds_Bs = (unsigned)(unsigned long long)&Bs[0][0];

    const unsigned short* Ag = A  + (size_t)blockRow * lda;
    const unsigned short* Bg = Wt + (size_t)blockCol * K;

    v8f acc[2][4];
#pragma unroll
    for (int mb = 0; mb < 2; ++mb)
#pragma unroll
        for (int nb = 0; nb < 4; ++nb) acc[mb][nb] = (v8f){};

    const int nk = K / BK;

    // preload tile 0
    if (wave == 0) {
        tdm_load_2d(lds_As, Ag, BK, BM, (unsigned long long)lda, (unsigned)K, (unsigned)M);
        tdm_load_2d(lds_Bs, Bg, BK, BN, (unsigned long long)K,   (unsigned)K, (unsigned)N);
    }

    for (int kt = 0; kt < nk; ++kt) {
        const int cur = kt & 1;

        if (wave == 0) {
            if (kt + 1 < nk) {
                const int k1 = (kt + 1) * BK;
                tdm_load_2d(lds_As + (1 - cur) * TILE_BYTES, Ag + k1,
                            BK, BM, (unsigned long long)lda, (unsigned)K, (unsigned)M);
                tdm_load_2d(lds_Bs + (1 - cur) * TILE_BYTES, Bg + k1,
                            BK, BN, (unsigned long long)K,   (unsigned)K, (unsigned)N);
                __builtin_amdgcn_s_wait_tensorcnt(2);   // current pair done
            } else {
                __builtin_amdgcn_s_wait_tensorcnt(0);
            }
        }
        __syncthreads();

        const unsigned short* Ab = As[cur];
        const unsigned short* Bb = Bs[cur];

        // ---- fragments ----
        AFrag af[2];
#pragma unroll
        for (int mb = 0; mb < 2; ++mb) {
            int row = wm0 + mb * 16 + ml;
            const unsigned int* p = (const unsigned int*)&Ab[row * BK];
#pragma unroll
            for (int v = 0; v < 8; ++v) {
                int kb = ((v < 4) ? (2 * v) : (16 + 2 * (v - 4))) + half * 8;
                af[mb].u[v] = p[kb >> 1];
            }
        }
        AFrag bfr[4];
#pragma unroll
        for (int nb = 0; nb < 4; ++nb) {
            int col = wn0 + nb * 16 + ml;
            const unsigned int* p = (const unsigned int*)&Bb[col * BK];
#pragma unroll
            for (int v = 0; v < 8; ++v) {
                int kk = ((v < 4) ? (2 * v) : (16 + 2 * (v - 4))) + half * 8;
                bfr[nb].u[v] = p[kk >> 1];
            }
        }
#pragma unroll
        for (int mb = 0; mb < 2; ++mb)
#pragma unroll
            for (int nb = 0; nb < 4; ++nb)
                acc[mb][nb] = wmma_bf16(af[mb].v, bfr[nb].v, acc[mb][nb]);

        __syncthreads();
    }

    // ---- epilogue ----
#pragma unroll
    for (int mb = 0; mb < 2; ++mb) {
#pragma unroll
        for (int nb = 0; nb < 4; ++nb) {
            int col = blockCol + wn0 + nb * 16 + ml;
            float bv = bias ? bias[col] : 0.0f;
#pragma unroll
            for (int r = 0; r < 8; ++r) {
                int row = blockRow + wm0 + mb * 16 + 8 * half + r;
                float v = acc[mb][nb][r] + bv;
                if (do_gelu) v = gelu_tanh(v);
                size_t oi = (size_t)row * ldc + col;
                if (residual) v += residual[oi];
                if (outf) outf[oi] = v;
                if (outb) outb[oi] = f2bf(v);
            }
        }
    }
}

// ---------------------------------------------------------------------------
// Flash attention: one wave per 16-query tile per (b,h). D=64 fixed.
// K/V 32x64 tiles are double-buffered via the Tensor Data Mover, then
// S = Q K^T (WMMA), online softmax, O += P V (WMMA). bf16 in / bf16 out.
// ---------------------------------------------------------------------------
__global__ __launch_bounds__(32) void flash_attn_kernel(
    const unsigned short* __restrict__ Q, int qstride,
    const unsigned short* __restrict__ Kt, int kstride,
    const unsigned short* __restrict__ V, int vstride,
    unsigned short* __restrict__ O, int ostride,
    int Lq, int Lkv, float scale) {

    const int b  = blockIdx.z;
    const int h  = blockIdx.y;
    const int q0 = blockIdx.x * 16;

    const int lane = threadIdx.x;
    const int half = lane >> 4;
    const int ml   = lane & 15;

    __shared__ unsigned short Ks[2][32 * 64];   // K tiles (TDM dest)
    __shared__ unsigned short Vs[2][32 * 64];   // V tiles (TDM dest)
    __shared__ unsigned short Ps[16 * 32];      // P tile, bf16

    const unsigned lds_Ks = (unsigned)(unsigned long long)&Ks[0][0];
    const unsigned lds_Vs = (unsigned)(unsigned long long)&Vs[0][0];
    const unsigned kvTileBytes = 32 * 64 * 2;   // 4096

    const unsigned short* Kg = Kt + (size_t)b * Lkv * kstride + h * 64;
    const unsigned short* Vg = V  + (size_t)b * Lkv * vstride + h * 64;

    // ---- load Q fragments (16 rows x 64 d -> two k=32 A-fragments) ----
    AFrag aq[2];
    {
        const unsigned short* qrow = Q + ((size_t)(b * Lq + q0 + ml)) * qstride + h * 64;
#pragma unroll
        for (int c = 0; c < 2; ++c)
#pragma unroll
            for (int v = 0; v < 8; ++v) {
                int kb = ((v < 4) ? (2 * v) : (16 + 2 * (v - 4))) + half * 8 + c * 32;
                aq[c].u[v] = *(const unsigned int*)(qrow + kb);
            }
    }

    float mrow[8], lrow[8];
#pragma unroll
    for (int r = 0; r < 8; ++r) { mrow[r] = -1e30f; lrow[r] = 0.0f; }
    v8f oacc[4];
#pragma unroll
    for (int nb = 0; nb < 4; ++nb) oacc[nb] = (v8f){};

    const int ntiles = Lkv / 32;

    // preload tile 0
    tdm_load_2d(lds_Ks, Kg, 64, 32, (unsigned long long)kstride, 64, 32);
    tdm_load_2d(lds_Vs, Vg, 64, 32, (unsigned long long)vstride, 64, 32);

    for (int t = 0; t < ntiles; ++t) {
        const int cur = t & 1;

        if (t + 1 < ntiles) {
            const size_t kb1 = (size_t)(t + 1) * 32;
            tdm_load_2d(lds_Ks + (1 - cur) * kvTileBytes, Kg + kb1 * kstride,
                        64, 32, (unsigned long long)kstride, 64, 32);
            tdm_load_2d(lds_Vs + (1 - cur) * kvTileBytes, Vg + kb1 * vstride,
                        64, 32, (unsigned long long)vstride, 64, 32);
            __builtin_amdgcn_s_wait_tensorcnt(2);   // current pair done
        } else {
            __builtin_amdgcn_s_wait_tensorcnt(0);
        }
        __syncthreads();

        // ---- S = Q K^T for this 32-key tile: two 16x16 f32 accumulators ----
        v8f s[2];
#pragma unroll
        for (int kv = 0; kv < 2; ++kv) {
            AFrag bk0, bk1;
            const unsigned short* krow = &Ks[cur][(kv * 16 + ml) * 64];
#pragma unroll
            for (int v = 0; v < 8; ++v) {
                int d = ((v < 4) ? (2 * v) : (16 + 2 * (v - 4))) + half * 8;
                bk0.u[v] = *(const unsigned int*)(krow + d);
                bk1.u[v] = *(const unsigned int*)(krow + 32 + d);
            }
            v8f z = (v8f){};
            z = wmma_bf16(aq[0].v, bk0.v, z);
            z = wmma_bf16(aq[1].v, bk1.v, z);
            s[kv] = z;
        }

        // ---- online softmax, per owned row r (global row = r + 8*half) ----
#pragma unroll
        for (int r = 0; r < 8; ++r) {
            float s0 = s[0][r] * scale;
            float s1 = s[1][r] * scale;
            float tmax = fmaxf(s0, s1);
#pragma unroll
            for (int off = 8; off; off >>= 1)
                tmax = fmaxf(tmax, __shfl_xor(tmax, off, 32));
            float newm = fmaxf(mrow[r], tmax);
            float p0 = __expf(s0 - newm);
            float p1 = __expf(s1 - newm);
            float psum = p0 + p1;
#pragma unroll
            for (int off = 8; off; off >>= 1)
                psum += __shfl_xor(psum, off, 32);
            float corr = __expf(mrow[r] - newm);
            lrow[r] = lrow[r] * corr + psum;
            mrow[r] = newm;
#pragma unroll
            for (int nb = 0; nb < 4; ++nb) oacc[nb][r] *= corr;

            int prow = r + 8 * half;
            Ps[prow * 32 + ml]      = f2bf(p0);
            Ps[prow * 32 + 16 + ml] = f2bf(p1);
        }
        __syncthreads();

        // ---- repack P(16x32) into an A-fragment ----
        AFrag ap;
        {
            const unsigned int* pp = (const unsigned int*)&Ps[ml * 32];
#pragma unroll
            for (int v = 0; v < 8; ++v) {
                int kb = ((v < 4) ? (2 * v) : (16 + 2 * (v - 4))) + half * 8;
                ap.u[v] = pp[kb >> 1];
            }
        }

        // ---- O += P V (V read from LDS tile) ----
#pragma unroll
        for (int nb = 0; nb < 4; ++nb) {
            AFrag bv;
            int dcol = nb * 16 + ml;
#pragma unroll
            for (int v = 0; v < 8; ++v) {
                int keyl = ((v < 4) ? (2 * v) : (16 + 2 * (v - 4))) + half * 8;
                const unsigned short* vr = &Vs[cur][keyl * 64 + dcol];
                bv.u[v] = ((unsigned int)vr[0]) | (((unsigned int)vr[64]) << 16);
            }
            oacc[nb] = wmma_bf16(ap.v, bv.v, oacc[nb]);
        }
        __syncthreads();
    }

    // ---- normalize and write bf16 output (b, row, h*64 + d) ----
#pragma unroll
    for (int nb = 0; nb < 4; ++nb)
#pragma unroll
        for (int r = 0; r < 8; ++r) {
            int row = q0 + r + 8 * half;
            float val = oacc[nb][r] / lrow[r];
            O[((size_t)(b * Lq + row)) * ostride + h * 64 + nb * 16 + ml] = f2bf(val);
        }
}

// ---------------------------------------------------------------------------
// Host orchestration
// ---------------------------------------------------------------------------
extern "C" void kernel_launch(void* const* d_in, const int* in_sizes, int n_in,
                              void* d_out, int out_size, void* d_ws, size_t ws_size,
                              hipStream_t stream) {
    (void)in_sizes; (void)n_in; (void)out_size; (void)ws_size;

    const int B = 4, L = 2048, C = 1024, Lc = 1024, H = 16, Dff = 4096;
    const int BL = B * L;       // 8192
    const int BLc = B * Lc;     // 4096

    const float* x       = (const float*)d_in[0];
    const float* context = (const float*)d_in[1];
    const float* w_qkv = (const float*)d_in[2];  const float* b_qkv = (const float*)d_in[3];
    const float* w_os  = (const float*)d_in[4];  const float* b_os  = (const float*)d_in[5];
    const float* w_qc  = (const float*)d_in[6];  const float* b_qc  = (const float*)d_in[7];
    const float* w_kvc = (const float*)d_in[8];  const float* b_kvc = (const float*)d_in[9];
    const float* w_oc  = (const float*)d_in[10]; const float* b_oc  = (const float*)d_in[11];
    const float* w_fc1 = (const float*)d_in[12]; const float* b_fc1 = (const float*)d_in[13];
    const float* w_fc2 = (const float*)d_in[14]; const float* b_fc2 = (const float*)d_in[15];

    float* out = (float*)d_out;
    char*  ws  = (char*)d_ws;

    // bump allocator over workspace
    size_t off = 0;
    auto alloc = [&](size_t bytes) -> char* {
        char* p = ws + off;
        off += (bytes + 255) & ~(size_t)255;
        return p;
    };

    unsigned short* wqkv_bf = (unsigned short*)alloc((size_t)C * 3 * C * 2);  // transposed
    unsigned short* wos_bf  = (unsigned short*)alloc((size_t)C * C * 2);
    unsigned short* wqc_bf  = (unsigned short*)alloc((size_t)C * C * 2);
    unsigned short* wkvc_bf = (unsigned short*)alloc((size_t)C * 2 * C * 2);
    unsigned short* woc_bf  = (unsigned short*)alloc((size_t)C * C * 2);
    unsigned short* wfc1_bf = (unsigned short*)alloc((size_t)C * Dff * 2);
    unsigned short* wfc2_bf = (unsigned short*)alloc((size_t)Dff * C * 2);
    unsigned short* ctx_bf  = (unsigned short*)alloc((size_t)BLc * C * 2);
    unsigned short* buf_ln  = (unsigned short*)alloc((size_t)BL * C * 2);
    unsigned short* buf_big = (unsigned short*)alloc((size_t)BL * Dff * 2); // qkv / kv / ffn-mid
    unsigned short* buf_q   = (unsigned short*)alloc((size_t)BL * C * 2);
    unsigned short* buf_at  = (unsigned short*)alloc((size_t)BL * C * 2);
    float*          x1      = (float*)alloc((size_t)BL * C * 4);

    // W (KxN f32) -> W^T (NxK bf16)
    auto cvtT = [&](const float* src, unsigned short* dst, int K, int N) {
        dim3 g(K / 32, N / 32);
        cvtT_f32_bf16_kernel<<<g, 256, 0, stream>>>(src, dst, K, N);
    };
    auto cvt = [&](const float* src, unsigned short* dst, size_t n) {
        cvt_f32_bf16_kernel<<<2048, 256, 0, stream>>>(src, dst, n);
    };
    auto gemm = [&](const unsigned short* A, int lda, const unsigned short* WtP,
                    const float* bias, const float* resid, float* of, unsigned short* ob,
                    int M, int N, int K, int do_gelu) {
        dim3 g(M / BM, N / BN);
        gemm_bf16_kernel<<<g, 256, 0, stream>>>(A, lda, WtP, bias, resid, of, ob,
                                                M, N, K, N, do_gelu);
    };

    // --- weight transpose+convert, context convert (deterministic every call) ---
    cvtT(w_qkv, wqkv_bf, C, 3 * C);
    cvtT(w_os,  wos_bf,  C, C);
    cvtT(w_qc,  wqc_bf,  C, C);
    cvtT(w_kvc, wkvc_bf, C, 2 * C);
    cvtT(w_oc,  woc_bf,  C, C);
    cvtT(w_fc1, wfc1_bf, C, Dff);
    cvtT(w_fc2, wfc2_bf, Dff, C);
    cvt(context, ctx_bf, (size_t)BLc * C);

    const float scale = 0.125f;  // 1/sqrt(64)

    // --- self-attention ---
    ln_bf16_kernel<<<BL, 256, 0, stream>>>(x, buf_ln);
    gemm(buf_ln, C, wqkv_bf, b_qkv, nullptr, nullptr, buf_big, BL, 3 * C, C, 0);
    {
        dim3 g(L / 16, H, B);
        flash_attn_kernel<<<g, 32, 0, stream>>>(buf_big,          3 * C,
                                                buf_big + C,      3 * C,
                                                buf_big + 2 * C,  3 * C,
                                                buf_at, C, L, L, scale);
    }
    gemm(buf_at, C, wos_bf, b_os, x, x1, nullptr, BL, C, C, 0);

    // --- cross-attention ---
    ln_bf16_kernel<<<BL, 256, 0, stream>>>(x1, buf_ln);
    gemm(buf_ln, C, wqc_bf, b_qc, nullptr, nullptr, buf_q, BL, C, C, 0);
    gemm(ctx_bf, C, wkvc_bf, b_kvc, nullptr, nullptr, buf_big, BLc, 2 * C, C, 0);
    {
        dim3 g(L / 16, H, B);
        flash_attn_kernel<<<g, 32, 0, stream>>>(buf_q,        C,
                                                buf_big,      2 * C,
                                                buf_big + C,  2 * C,
                                                buf_at, C, L, Lc, scale);
    }
    gemm(buf_at, C, woc_bf, b_oc, x1, out, nullptr, BL, C, C, 0);  // out = x2

    // --- FFN ---
    ln_bf16_kernel<<<BL, 256, 0, stream>>>(out, buf_ln);
    gemm(buf_ln, C, wfc1_bf, b_fc1, nullptr, nullptr, buf_big, BL, Dff, C, 1);
    gemm(buf_big, Dff, wfc2_bf, b_fc2, out, out, nullptr, BL, C, Dff, 0);
}
